// DecomposingAttnProcessor_pad_41669772706195
// MI455X (gfx1250) — compile-verified, matrix operand order checked by reading, not run on hardware
//
#include <hip/hip_runtime.h>
#include <math.h>

typedef __bf16 bf16;
typedef __attribute__((ext_vector_type(16))) __bf16 v16bf;
typedef __attribute__((ext_vector_type(8)))  float  v8f;

#define HEADS 24
#define DIM_HEAD 64
#define DMODEL 1536
#define CROSS 2048
#define SEQ 2048
#define EKV 77
#define NCOMP 4
#define BATCH 2
#define MKV_PAD 640

// ---------------- CDNA5 async global->LDS DMA helpers (inline asm, ASYNCcnt) ----------------
// LDS byte offset = low 32 bits of the generic pointer (ISA 10.2: LDS aperture keeps the
// wave-relative LDS address in addr[31:0]).
__device__ __forceinline__ void async_copy16(void* lds_dst, const void* gsrc) {
  unsigned loff = (unsigned)(uintptr_t)lds_dst;
  asm volatile("global_load_async_to_lds_b128 %0, %1, off" ::"v"(loff), "v"(gsrc)
               : "memory");
}
#define WAIT_ASYNC_0() asm volatile("s_wait_asynccnt 0x0" ::: "memory")
#define WAIT_ASYNC_4() asm volatile("s_wait_asynccnt 0x4" ::: "memory")

__device__ __forceinline__ v8f vzero8() {
  v8f z;
#pragma unroll
  for (int e = 0; e < 8; ++e) z[e] = 0.0f;
  return z;
}

// K-index inside a 16x32 (MxK) bf16 A-fragment / B^T-fragment for element e, lane l.
// ISA 7.12.2: lanes 0-15 hold K {0..7,16..23}; lanes 16-31 hold K {8..15,24..31}.
__device__ __forceinline__ int frag_k(int e, int lane) {
  return ((e < 8) ? e : (e + 8)) + ((lane < 16) ? 0 : 8);
}

// Load a 16x32 row-major tile (row = M or N index, col = K) from LDS into a fragment.
__device__ __forceinline__ v16bf load_frag(const bf16* base, int row, int stride,
                                           int koff, int lane) {
  v16bf f;
  const bf16* p = base + row * stride + koff;
#pragma unroll
  for (int e = 0; e < 16; ++e) f[e] = p[frag_k(e, lane)];
  return f;
}

// Load a B fragment from a K-major (rows = K, cols = N) LDS tile.
__device__ __forceinline__ v16bf load_frag_kmajor(const bf16* base, int ncol,
                                                  int stride, int lane) {
  v16bf f;
#pragma unroll
  for (int e = 0; e < 16; ++e) f[e] = base[frag_k(e, lane) * stride + ncol];
  return f;
}

__device__ __forceinline__ v8f wmma_bf16(v16bf a, v16bf b, v8f c) {
  return __builtin_amdgcn_wmma_f32_16x16x32_bf16(false, a, false, b, (short)0, c,
                                                 false, false);
}

// ---------------- f32 -> bf16 conversion with zero tail padding ----------------
__global__ __launch_bounds__(256) void cvt_kernel(const float* __restrict__ x,
                                                  bf16* __restrict__ y, long n_src,
                                                  long n_total) {
  long i = ((long)blockIdx.x * blockDim.x + threadIdx.x) * 8;
  if (i >= n_total) return;
  bf16 tmp[8];
  if (i < n_src) {
#pragma unroll
    for (int e = 0; e < 8; ++e) tmp[e] = (bf16)x[i + e];
  } else {
#pragma unroll
    for (int e = 0; e < 8; ++e) tmp[e] = (bf16)0.0f;
  }
  uint4 pk;
  __builtin_memcpy(&pk, tmp, 16);
  *(uint4*)(y + i) = pk;
}

// -------------------- Projection GEMM: Y = X @ W^T + bias --------------------
// X: [M,K] bf16, W: [N,K] bf16, Y: [M,N] (bf16 or f32). 128x128 block, K-step 32,
// double-buffered LDS filled by async global->LDS DMA. M, N multiples of 128.
template <typename OT>
__global__ __launch_bounds__(256) void proj_gemm(const bf16* __restrict__ X,
                                                 const bf16* __restrict__ W,
                                                 const float* __restrict__ bias,
                                                 OT* __restrict__ Y, int M, int N,
                                                 int K) {
  __shared__ __align__(16) bf16 As[2][128 * 40];
  __shared__ __align__(16) bf16 Bs[2][128 * 40];
  const int tid = threadIdx.x;
  const int lane = tid & 31;
  const int lm = lane & 15;
  const int half = lane >> 4;
  const int wave = tid >> 5;
  const int m0 = blockIdx.y * 128;
  const int n0 = blockIdx.x * 128;
  const int wm = (wave & 3) * 32;   // 4 waves along M (32 rows each)
  const int wn = (wave >> 2) * 64;  // 2 waves along N (64 cols each)

  // This thread's two 16-byte chunks of each 128x32 tile (fixed across K-steps).
  const int r0 = tid >> 2, r1 = 64 + (tid >> 2);
  const int c8 = (tid & 3) * 8;

  auto issue = [&](int buf, int k0) {
    async_copy16(&As[buf][r0 * 40 + c8], X + (size_t)(m0 + r0) * K + k0 + c8);
    async_copy16(&Bs[buf][r0 * 40 + c8], W + (size_t)(n0 + r0) * K + k0 + c8);
    async_copy16(&As[buf][r1 * 40 + c8], X + (size_t)(m0 + r1) * K + k0 + c8);
    async_copy16(&Bs[buf][r1 * 40 + c8], W + (size_t)(n0 + r1) * K + k0 + c8);
  };

  v8f acc[2][4];
#pragma unroll
  for (int mi = 0; mi < 2; ++mi)
#pragma unroll
    for (int ni = 0; ni < 4; ++ni) acc[mi][ni] = vzero8();

  const int KT = K >> 5;
  issue(0, 0);
  for (int kt = 0; kt < KT; ++kt) {
    const int buf = kt & 1;
    if (kt + 1 < KT) {
      issue(buf ^ 1, (kt + 1) * 32);
      WAIT_ASYNC_4();  // own 4 DMAs for current buffer complete (in-order)
    } else {
      WAIT_ASYNC_0();
    }
    __syncthreads();  // everyone's DMAs for this buffer complete

    v16bf af[2], fb[4];
#pragma unroll
    for (int mi = 0; mi < 2; ++mi)
      af[mi] = load_frag(&As[buf][0], wm + mi * 16 + lm, 40, 0, lane);
#pragma unroll
    for (int ni = 0; ni < 4; ++ni)
      fb[ni] = load_frag(&Bs[buf][0], wn + ni * 16 + lm, 40, 0, lane);
#pragma unroll
    for (int mi = 0; mi < 2; ++mi)
#pragma unroll
      for (int ni = 0; ni < 4; ++ni)
        acc[mi][ni] = wmma_bf16(af[mi], fb[ni], acc[mi][ni]);
    __syncthreads();  // reads done before this buffer is overwritten
  }

#pragma unroll
  for (int mi = 0; mi < 2; ++mi) {
#pragma unroll
    for (int ni = 0; ni < 4; ++ni) {
      int n = n0 + wn + ni * 16 + lm;
      float bv = bias[n];
#pragma unroll
      for (int g = 0; g < 8; ++g) {
        int m = m0 + wm + mi * 16 + g + half * 8;
        Y[(size_t)m * N + n] = (OT)(acc[mi][ni][g] + bv);
      }
    }
  }
}

// -------------------- Fused decomposing attention --------------------
// grid: (S/16, HEADS, BATCH), 1 wave per block. Handles all 4 components for a
// 16-row q tile; softmax runs across components elementwise in registers.
__global__ __launch_bounds__(32) void attn_kernel(const bf16* __restrict__ q,
                                                  const bf16* __restrict__ k,
                                                  const bf16* __restrict__ v,
                                                  const float* __restrict__ bk,
                                                  const float* __restrict__ bvb,
                                                  bf16* __restrict__ attn_out) {
  const int i0 = blockIdx.x * 16;
  const int h = blockIdx.y;
  const int b = blockIdx.z;
  const int lane = threadIdx.x & 31;
  const int lm = lane & 15;
  const int half = lane >> 4;
  const float scale = 0.125f;  // 1/sqrt(64)

  __shared__ __align__(16) bf16 qs[NCOMP][16 * 72];
  __shared__ __align__(16) bf16 ks[NCOMP][32 * 72];
  __shared__ __align__(16) bf16 vs[NCOMP][32 * 72];
  __shared__ __align__(16) bf16 wb[16 * 40];
  __shared__ float sp_s[NCOMP][16];
  __shared__ float sw_s[NCOMP][16];

  // stage q tiles (16 x 64 per component) into LDS via async DMA
#pragma unroll
  for (int c = 0; c < NCOMP; ++c) {
    if (lane < 16) {
      const bf16* qg =
          q + ((size_t)(c * BATCH + b) * SEQ + (i0 + lane)) * DMODEL + h * DIM_HEAD;
      bf16* qd = &qs[c][lane * 72];
#pragma unroll
      for (int u = 0; u < 8; ++u) async_copy16(qd + u * 8, qg + u * 8);
    }
  }
  WAIT_ASYNC_0();
  __syncthreads();

  v16bf qf[NCOMP][2];
#pragma unroll
  for (int c = 0; c < NCOMP; ++c)
#pragma unroll
    for (int s = 0; s < 2; ++s) qf[c][s] = load_frag(&qs[c][0], lm, 72, s * 32, lane);

  // pad-branch scalar: sp = softmax_C( scale * (q_row . bk_head) )
  const float* bkh = bk + h * DIM_HEAD;
  float qb[NCOMP];
#pragma unroll
  for (int c = 0; c < NCOMP; ++c) {
    float p = 0.0f;
#pragma unroll
    for (int s = 0; s < 2; ++s)
#pragma unroll
      for (int e = 0; e < 16; ++e)
        p += (float)qf[c][s][e] * bkh[s * 32 + frag_k(e, lane)];
    p += __shfl_xor(p, 16, 32);  // combine both K halves for row m = lm
    qb[c] = p * scale;
  }
  {
    float mx = fmaxf(fmaxf(qb[0], qb[1]), fmaxf(qb[2], qb[3]));
    float e0 = __expf(qb[0] - mx), e1 = __expf(qb[1] - mx);
    float e2 = __expf(qb[2] - mx), e3 = __expf(qb[3] - mx);
    float inv = 1.0f / (e0 + e1 + e2 + e3);
    if (lane < 16) {
      sp_s[0][lane] = e0 * inv;
      sp_s[1][lane] = e1 * inv;
      sp_s[2][lane] = e2 * inv;
      sp_s[3][lane] = e3 * inv;
    }
  }

  v8f o[NCOMP][4];
#pragma unroll
  for (int c = 0; c < NCOMP; ++c)
#pragma unroll
    for (int t = 0; t < 4; ++t) o[c][t] = vzero8();
  float swp[NCOMP][8];
#pragma unroll
  for (int c = 0; c < NCOMP; ++c)
#pragma unroll
    for (int g = 0; g < 8; ++g) swp[c][g] = 0.0f;

  for (int jt = 0; jt < 3; ++jt) {
    const int j0 = jt * 32;
    __syncthreads();
    // 32-row k/v tiles via async DMA; rows >= 77 zero-filled with DS stores
#pragma unroll
    for (int c = 0; c < NCOMP; ++c) {
      int bc = c * BATCH + b;
      int jg = j0 + lane;
      bf16* kd = &ks[c][lane * 72];
      bf16* vd = &vs[c][lane * 72];
      if (jg < EKV) {
        const bf16* kg = k + ((size_t)(bc * EKV + jg)) * DMODEL + h * DIM_HEAD;
        const bf16* vg = v + ((size_t)(bc * EKV + jg)) * DMODEL + h * DIM_HEAD;
#pragma unroll
        for (int u = 0; u < 8; ++u) {
          async_copy16(kd + u * 8, kg + u * 8);
          async_copy16(vd + u * 8, vg + u * 8);
        }
      } else {
        uint4 z = {0u, 0u, 0u, 0u};
#pragma unroll
        for (int u = 0; u < 8; ++u) {
          *(uint4*)(kd + u * 8) = z;
          *(uint4*)(vd + u * 8) = z;
        }
      }
    }
    WAIT_ASYNC_0();
    __syncthreads();

    // scores for all 4 components: S_c = q_c (16x64) @ k_c^T (64x32)
    v8f sc[NCOMP][2];
#pragma unroll
    for (int c = 0; c < NCOMP; ++c) {
      sc[c][0] = vzero8();
      sc[c][1] = vzero8();
#pragma unroll
      for (int s = 0; s < 2; ++s) {
        v16bf b0 = load_frag(&ks[c][0], lm, 72, s * 32, lane);       // j sub-tile 0
        v16bf b1 = load_frag(&ks[c][0], 16 + lm, 72, s * 32, lane);  // j sub-tile 1
        sc[c][0] = wmma_bf16(qf[c][s], b0, sc[c][0]);
        sc[c][1] = wmma_bf16(qf[c][s], b1, sc[c][1]);
      }
    }

    // softmax across components (pure elementwise on matching fragments)
#pragma unroll
    for (int sub = 0; sub < 2; ++sub) {
      int j = j0 + sub * 16 + lm;
      float valid = (j < EKV) ? 1.0f : 0.0f;
#pragma unroll
      for (int g = 0; g < 8; ++g) {
        float x0 = scale * sc[0][sub][g], x1 = scale * sc[1][sub][g];
        float x2 = scale * sc[2][sub][g], x3 = scale * sc[3][sub][g];
        float mx = fmaxf(fmaxf(x0, x1), fmaxf(x2, x3));
        float e0 = __expf(x0 - mx), e1 = __expf(x1 - mx);
        float e2 = __expf(x2 - mx), e3 = __expf(x3 - mx);
        float inv = valid / (e0 + e1 + e2 + e3);
        sc[0][sub][g] = e0 * inv;
        sc[1][sub][g] = e1 * inv;
        sc[2][sub][g] = e2 * inv;
        sc[3][sub][g] = e3 * inv;
        swp[0][g] += sc[0][sub][g];
        swp[1][g] += sc[1][sub][g];
        swp[2][g] += sc[2][sub][g];
        swp[3][g] += sc[3][sub][g];
      }
    }

    // repack weights (C-layout) -> A-fragment layout via LDS, then O += W @ V
#pragma unroll
    for (int c = 0; c < NCOMP; ++c) {
      __syncthreads();
#pragma unroll
      for (int sub = 0; sub < 2; ++sub)
#pragma unroll
        for (int g = 0; g < 8; ++g)
          wb[(g + half * 8) * 40 + sub * 16 + lm] = (bf16)sc[c][sub][g];
      __syncthreads();
      v16bf aw = load_frag(wb, lm, 40, 0, lane);
#pragma unroll
      for (int t = 0; t < 4; ++t) {
        v16bf vb = load_frag_kmajor(&vs[c][0], t * 16 + lm, 72, lane);
        o[c][t] = wmma_bf16(aw, vb, o[c][t]);
      }
    }
  }

  // row sums of w_main (reduce the 16-lane halves; lanes 0-15 own rows g, 16-31 rows g+8)
#pragma unroll
  for (int c = 0; c < NCOMP; ++c)
#pragma unroll
    for (int g = 0; g < 8; ++g) {
      float s = swp[c][g];
      s += __shfl_xor(s, 1, 32);
      s += __shfl_xor(s, 2, 32);
      s += __shfl_xor(s, 4, 32);
      s += __shfl_xor(s, 8, 32);
      if (lm == 0) sw_s[c][g + half * 8] = s;
    }
  __syncthreads();

  // epilogue: O + pad-branch rank-1 correction, store bf16
  const float cpad = (float)NCOMP / (float)EKV;
  const float* bvh = bvb + h * DIM_HEAD;
#pragma unroll
  for (int c = 0; c < NCOMP; ++c) {
    int bc = c * BATCH + b;
#pragma unroll
    for (int t = 0; t < 4; ++t) {
      int d = t * 16 + lm;
      float bvv = bvh[d];
#pragma unroll
      for (int g = 0; g < 8; ++g) {
        int m = g + half * 8;
        float padw = cpad * sp_s[c][m] * ((float)EKV - sw_s[c][m]);
        float val = o[c][t][g] + padw * bvv;
        attn_out[((size_t)bc * SEQ + (i0 + m)) * DMODEL + (h * DIM_HEAD + d)] = (bf16)val;
      }
    }
  }
}

extern "C" void kernel_launch(void* const* d_in, const int* in_sizes, int n_in,
                              void* d_out, int out_size, void* d_ws, size_t ws_size,
                              hipStream_t stream) {
  const float* hs = (const float*)d_in[0];
  const float* enc = (const float*)d_in[1];
  const float* Wq = (const float*)d_in[2];
  const float* bq = (const float*)d_in[3];
  const float* Wk = (const float*)d_in[4];
  const float* bk = (const float*)d_in[5];
  const float* Wv = (const float*)d_in[6];
  const float* bv = (const float*)d_in[7];
  const float* Wo = (const float*)d_in[8];
  const float* bo = (const float*)d_in[9];
  float* out = (float*)d_out;

  const long MQ = (long)NCOMP * BATCH * SEQ;   // 16384
  const long MKV = (long)NCOMP * BATCH * EKV;  // 616

  // workspace carve-up (all bf16)
  bf16* hs_bf = (bf16*)d_ws;                                // 16384x1536
  bf16* enc_bf = hs_bf + MQ * DMODEL;                       // 640x2048 (zero-padded)
  bf16* wq_bf = enc_bf + (long)MKV_PAD * CROSS;             // 1536x1536
  bf16* wk_bf = wq_bf + (long)DMODEL * DMODEL;              // 1536x2048
  bf16* wv_bf = wk_bf + (long)DMODEL * CROSS;               // 1536x2048
  bf16* wo_bf = wv_bf + (long)DMODEL * CROSS;               // 1536x1536
  bf16* qbuf = wo_bf + (long)DMODEL * DMODEL;               // 16384x1536
  bf16* kbuf = qbuf + MQ * DMODEL;                          // 640x1536
  bf16* vbuf = kbuf + (long)MKV_PAD * DMODEL;               // 640x1536
  bf16* abuf = vbuf + (long)MKV_PAD * DMODEL;               // 16384x1536

  auto cvt = [&](const float* x, bf16* y, long n_src, long n_total) {
    long blocks = (n_total / 8 + 255) / 256;
    cvt_kernel<<<dim3((unsigned)blocks), dim3(256), 0, stream>>>(x, y, n_src, n_total);
  };
  cvt(hs, hs_bf, MQ * DMODEL, MQ * DMODEL);
  cvt(enc, enc_bf, MKV * CROSS, (long)MKV_PAD * CROSS);
  cvt(Wq, wq_bf, (long)DMODEL * DMODEL, (long)DMODEL * DMODEL);
  cvt(Wk, wk_bf, (long)DMODEL * CROSS, (long)DMODEL * CROSS);
  cvt(Wv, wv_bf, (long)DMODEL * CROSS, (long)DMODEL * CROSS);
  cvt(Wo, wo_bf, (long)DMODEL * DMODEL, (long)DMODEL * DMODEL);

  dim3 blk(256);
  proj_gemm<bf16><<<dim3(DMODEL / 128, MQ / 128), blk, 0, stream>>>(
      hs_bf, wq_bf, bq, qbuf, (int)MQ, DMODEL, DMODEL);
  proj_gemm<bf16><<<dim3(DMODEL / 128, MKV_PAD / 128), blk, 0, stream>>>(
      enc_bf, wk_bf, bk, kbuf, MKV_PAD, DMODEL, CROSS);
  proj_gemm<bf16><<<dim3(DMODEL / 128, MKV_PAD / 128), blk, 0, stream>>>(
      enc_bf, wv_bf, bv, vbuf, MKV_PAD, DMODEL, CROSS);

  attn_kernel<<<dim3(SEQ / 16, HEADS, BATCH), dim3(32), 0, stream>>>(
      qbuf, kbuf, vbuf, bk, bv, abuf);

  proj_gemm<float><<<dim3(DMODEL / 128, MQ / 128), blk, 0, stream>>>(
      abuf, wo_bf, bo, out, (int)MQ, DMODEL, DMODEL);
}
